// NeuralSDEGenerator_74088185856495
// MI455X (gfx1250) — compile-verified, matrix operand order checked by reading, not run on hardware
//
#include <hip/hip_runtime.h>

// ---------------------------------------------------------------------------
// Neural SDE (reversible Heun) for MI455X / gfx1250 — cooperative version.
//   T=256 steps, B=4096, H=64, MLPW=128, D=8.
// One block (4 wave32) per 16 batch rows; the 4 waves split the feature
// dimension: each owns 16 state columns and 2 of 8 MLP n-tiles per layer.
// All matmuls are v_wmma_f32_16x16x32_f16; activations use v_exp_f32 +
// v_rcp_f32 (TRANS32 ops that co-execute with the XDL WMMA pipeline).
// ---------------------------------------------------------------------------

typedef _Float16 h8   __attribute__((ext_vector_type(8)));
typedef _Float16 v16h __attribute__((ext_vector_type(16)));
typedef float    v8f  __attribute__((ext_vector_type(8)));

#define T_STEPS 256
#define BATCH   4096
#define HID     64
#define MLPWID  128
#define OUTC    9       // 1 + DATA

// ws (f16) layout offsets, in halves
#define OFF_DR_W0T 0        // 128 x 64
#define OFF_DR_W1T 8192     // 128 x 128
#define OFF_DR_WOT 24576    // 64  x 128
#define OFF_DF_W0T 32768
#define OFF_DF_W1T 40960
#define OFF_DF_WOT 57344
#define OFF_ROT    65536    // 16 x 64 (rows 8..15 zero)
#define WS_HALVES  66560

__device__ __forceinline__ v16h pack16(h8 lo, h8 hi) {
  v16h r;
#pragma unroll
  for (int i = 0; i < 8; ++i) { r[i] = lo[i]; r[i + 8] = hi[i]; }
  return r;
}

// Load an A/B fragment: 8 halves at [row][k0..k0+7] and 8 at [row][k0+16..k0+23].
__device__ __forceinline__ v16h frag(const _Float16* base, int rowstride, int row, int k0) {
  const h8* p = (const h8*)(base + row * rowstride + k0);
  return pack16(p[0], p[2]);
}

__device__ __forceinline__ v8f wmma16(v16h a, v16h b, v8f c) {
  return __builtin_amdgcn_wmma_f32_16x16x32_f16(false, a, false, b, (short)0, c, false, false);
}

// Single v_rcp_f32 instead of the IEEE divide expansion (div_scale/fmas/fixup).
__device__ __forceinline__ float fast_rcp(float x) {
  return __builtin_amdgcn_rcpf(x);
}

__device__ __forceinline__ float lipswish_f(float x) {
  float e = __expf(-x);                      // v_exp_f32 (neg folds into modifier)
  return 0.909f * x * fast_rcp(1.0f + e);    // sigmoid via v_rcp_f32
}
__device__ __forceinline__ float tanh_f(float x) {
  float e = __expf(-2.0f * x);
  return (1.0f - e) * fast_rcp(1.0f + e);
}

struct MlpW {
  const _Float16* W0xT;  // 128 x 64   (n-major, k stride 64)
  const _Float16* W1T;   // 128 x 128
  const _Float16* WoT;   // 64  x 128
  const float* W0row0;   // original W0, row 0 = time column (128)
  const float* b0;       // 128
  const float* b1;       // 128
  const float* bo;       // 64
};

// Cooperative 3-layer MLP over 16 shared rows.
//   L1: wave computes n-tiles {2w, 2w+1} of h1   (2 x 2 WMMA)
//   L2: wave computes n-tiles {2w, 2w+1} of h2   (2 x 4 WMMA)
//   L3: wave computes n-tile w (its own 16 state columns) -> out[8]
// Caller guarantees xbuf is complete and h1 is writable on entry.
__device__ __forceinline__ void mlp_coop(const MlpW& w, float tval,
                                         const _Float16* xbuf,
                                         _Float16* h1, _Float16* h2,
                                         int lane, int wave, float (&out)[8]) {
  const int r  = lane & 15;
  const int hi = (lane >> 4) << 3;   // 0 or 8
  const int kb = hi;                 // k-base offset within fragment

  // ---- layer 1: (16x64) @ (64x128), bias' = b0 + t * W0[0,:]
  {
    v16h a0 = frag(xbuf, HID, r, 0 + kb);
    v16h a1 = frag(xbuf, HID, r, 32 + kb);
#pragma unroll
    for (int p = 0; p < 2; ++p) {
      const int col = (2 * wave + p) * 16 + r;
      const float bias = w.b0[col] + tval * w.W0row0[col];
      v8f acc;
#pragma unroll
      for (int i = 0; i < 8; ++i) acc[i] = bias;
      acc = wmma16(a0, frag(w.W0xT, HID, col, 0 + kb), acc);
      acc = wmma16(a1, frag(w.W0xT, HID, col, 32 + kb), acc);
#pragma unroll
      for (int i = 0; i < 8; ++i)
        h1[(i + hi) * MLPWID + col] = (_Float16)lipswish_f(acc[i]);
    }
  }
  __syncthreads();   // h1 complete

  // ---- layer 2: (16x128) @ (128x128)
  {
    v16h c0 = frag(h1, MLPWID, r, 0 + kb);
    v16h c1 = frag(h1, MLPWID, r, 32 + kb);
    v16h c2 = frag(h1, MLPWID, r, 64 + kb);
    v16h c3 = frag(h1, MLPWID, r, 96 + kb);
#pragma unroll
    for (int p = 0; p < 2; ++p) {
      const int col = (2 * wave + p) * 16 + r;
      const float bias = w.b1[col];
      v8f acc;
#pragma unroll
      for (int i = 0; i < 8; ++i) acc[i] = bias;
      acc = wmma16(c0, frag(w.W1T, MLPWID, col, 0 + kb), acc);
      acc = wmma16(c1, frag(w.W1T, MLPWID, col, 32 + kb), acc);
      acc = wmma16(c2, frag(w.W1T, MLPWID, col, 64 + kb), acc);
      acc = wmma16(c3, frag(w.W1T, MLPWID, col, 96 + kb), acc);
#pragma unroll
      for (int i = 0; i < 8; ++i)
        h2[(i + hi) * MLPWID + col] = (_Float16)lipswish_f(acc[i]);
    }
  }
  __syncthreads();   // h2 complete, h1 free

  // ---- layer 3: (16x128) @ (128x64) -> tanh, this wave's 16 columns only
  {
    v16h d0 = frag(h2, MLPWID, r, 0 + kb);
    v16h d1 = frag(h2, MLPWID, r, 32 + kb);
    v16h d2 = frag(h2, MLPWID, r, 64 + kb);
    v16h d3 = frag(h2, MLPWID, r, 96 + kb);
    const int col = wave * 16 + r;
    const float bias = w.bo[col];
    v8f acc;
#pragma unroll
    for (int i = 0; i < 8; ++i) acc[i] = bias;
    acc = wmma16(d0, frag(w.WoT, MLPWID, col, 0 + kb), acc);
    acc = wmma16(d1, frag(w.WoT, MLPWID, col, 32 + kb), acc);
    acc = wmma16(d2, frag(w.WoT, MLPWID, col, 64 + kb), acc);
    acc = wmma16(d3, frag(w.WoT, MLPWID, col, 96 + kb), acc);
#pragma unroll
    for (int i = 0; i < 8; ++i) out[i] = tanh_f(acc[i]);
  }
}

// This wave's state tile (C-layout f32) -> its 16 columns of the LDS f16 buffer.
__device__ __forceinline__ void store_tile_f16(_Float16* xbuf, const float (&s)[8],
                                               int lane, int wave) {
  const int r = lane & 15;
  const int hi = (lane >> 4) << 3;
#pragma unroll
  for (int i = 0; i < 8; ++i)
    xbuf[(i + hi) * HID + wave * 16 + r] = (_Float16)s[i];
}

// readout: (16x64)@(64x8 padded to 16) via WMMA; executed by wave 0 only.
__device__ __forceinline__ void readout_store(const _Float16* xbuf, const _Float16* roT,
                                              const float* readout_b, float tval, int t,
                                              int rowbase, int lane, float* out) {
  const int r = lane & 15;
  const int hi = (lane >> 4) << 3;
  const int kb = hi;
  v16h a0 = frag(xbuf, HID, r, 0 + kb);
  v16h a1 = frag(xbuf, HID, r, 32 + kb);
  const float bias = (r < 8) ? readout_b[r] : 0.0f;
  v8f acc;
#pragma unroll
  for (int i = 0; i < 8; ++i) acc[i] = bias;
  acc = wmma16(a0, frag(roT, HID, r, 0 + kb), acc);
  acc = wmma16(a1, frag(roT, HID, r, 32 + kb), acc);
#pragma unroll
  for (int i = 0; i < 8; ++i) {
    const int row = rowbase + i + hi;
    float* orow = out + ((size_t)row * T_STEPS + t) * OUTC;
    if (r < 8)       orow[1 + r] = acc[i];
    else if (r == 8) orow[0] = tval;
  }
}

// ---------------------------------------------------------------------------
// Prep: transpose + f16-convert all weight matrices into the workspace.
// ---------------------------------------------------------------------------
__global__ void sde_prep(const float* drW0, const float* drW1, const float* drWo,
                         const float* dfW0, const float* dfW1, const float* dfWo,
                         const float* roW, _Float16* wsh) {
  for (int i = blockIdx.x * blockDim.x + threadIdx.x; i < WS_HALVES;
       i += gridDim.x * blockDim.x) {
    _Float16 v;
    int j = i;
    if (j < OFF_DR_W1T)      { int n = j >> 6, k = j & 63;            v = (_Float16)drW0[(1 + k) * 128 + n]; }
    else if (j < OFF_DR_WOT) { j -= OFF_DR_W1T; int n = j >> 7, k = j & 127; v = (_Float16)drW1[k * 128 + n]; }
    else if (j < OFF_DF_W0T) { j -= OFF_DR_WOT; int n = j >> 7, k = j & 127; v = (_Float16)drWo[k * 64 + n]; }
    else if (j < OFF_DF_W1T) { j -= OFF_DF_W0T; int n = j >> 6, k = j & 63;  v = (_Float16)dfW0[(1 + k) * 128 + n]; }
    else if (j < OFF_DF_WOT) { j -= OFF_DF_W1T; int n = j >> 7, k = j & 127; v = (_Float16)dfW1[k * 128 + n]; }
    else if (j < OFF_ROT)    { j -= OFF_DF_WOT; int n = j >> 7, k = j & 127; v = (_Float16)dfWo[k * 64 + n]; }
    else                     { j -= OFF_ROT;    int n = j >> 6, k = j & 63;
                               v = (n < 8) ? (_Float16)roW[k * 8 + n] : (_Float16)0.0f; }
    wsh[i] = v;
  }
}

// ---------------------------------------------------------------------------
// Main kernel: one block (4 waves) per 16 batch rows; feature-split waves.
// ---------------------------------------------------------------------------
__global__ __launch_bounds__(128)
void sde_main(const float* ts, const float* dW,
              const float* drW0, const float* drb0, const float* drb1, const float* drbo,
              const float* dfW0, const float* dfb0, const float* dfb1, const float* dfbo,
              const float* rob, const _Float16* wsh, float* out) {
  __shared__ _Float16 xbuf[16 * HID];     // 2 KB : state, f16
  __shared__ _Float16 h1[16 * MLPWID];    // 4 KB
  __shared__ _Float16 h2[16 * MLPWID];    // 4 KB

  const int lane = threadIdx.x & 31;
  const int wave = threadIdx.x >> 5;      // 0..3 : owns state cols wave*16..+15
  const int rowbase = blockIdx.x * 16;
  const int r = lane & 15;
  const int hi = (lane >> 4) << 3;

  const MlpW drift = { wsh + OFF_DR_W0T, wsh + OFF_DR_W1T, wsh + OFF_DR_WOT,
                       drW0, drb0, drb1, drbo };
  const MlpW diffu = { wsh + OFF_DF_W0T, wsh + OFF_DF_W1T, wsh + OFF_DF_WOT,
                       dfW0, dfb0, dfb1, dfbo };
  const _Float16* roT = wsh + OFF_ROT;

  const float dt = ts[1] - ts[0];
  const float sq = sqrtf(dt);

  // This wave's 16-column state tile in C-layout: VGPR i is row i+hi,
  // lane column = wave*16 + (lane&15).
  float z[8], zh[8], f[8], g[8], u[8], dw[8];
#pragma unroll
  for (int i = 0; i < 8; ++i) { z[i] = 1.0f; zh[i] = 1.0f; }

  store_tile_f16(xbuf, z, lane, wave);    // x0 = ones
  __syncthreads();
  mlp_coop(drift, ts[0], xbuf, h1, h2, lane, wave, f);   // f0
  mlp_coop(diffu, ts[0], xbuf, h1, h2, lane, wave, g);   // g0
  __syncthreads();                        // diff-L3 h2 reads done; xbuf stable
  if (wave == 0)
    readout_store(xbuf, roT, rob, ts[0], 0, rowbase, lane, out);
  __syncthreads();                        // protect xbuf until readout done

#pragma unroll 1
  for (int t = 1; t < T_STEPS; ++t) {
    // Brownian increment dW[t-1] * sqrt(dt), this wave's columns.
    const float* dwp = dW + (size_t)(t - 1) * BATCH * HID;
#pragma unroll
    for (int i = 0; i < 8; ++i)
      dw[i] = sq * dwp[(size_t)(rowbase + i + hi) * HID + wave * 16 + r];

    // u = f*dt + g*dw ; zhat <- 2z - zhat + u
#pragma unroll
    for (int i = 0; i < 8; ++i) {
      u[i]  = f[i] * dt + g[i] * dw[i];
      zh[i] = 2.0f * z[i] - zh[i] + u[i];
    }

    store_tile_f16(xbuf, zh, lane, wave);
    __syncthreads();                      // xbuf(zhat) complete
    const float tv = ts[t];
    mlp_coop(drift, tv, xbuf, h1, h2, lane, wave, f);    // f_{n+1}
    mlp_coop(diffu, tv, xbuf, h1, h2, lane, wave, g);    // g_{n+1}

    // z <- z + 0.5*(u + f_{n+1}*dt + g_{n+1}*dw)
#pragma unroll
    for (int i = 0; i < 8; ++i)
      z[i] += 0.5f * (u[i] + f[i] * dt + g[i] * dw[i]);

    store_tile_f16(xbuf, z, lane, wave);
    __syncthreads();                      // xbuf(z) complete; h2 reads done
    if (wave == 0)
      readout_store(xbuf, roT, rob, tv, t, rowbase, lane, out);
    __syncthreads();                      // keep xbuf until readout done
  }
}

// ---------------------------------------------------------------------------
extern "C" void kernel_launch(void* const* d_in, const int* in_sizes, int n_in,
                              void* d_out, int out_size, void* d_ws, size_t ws_size,
                              hipStream_t stream) {
  const float* ts   = (const float*)d_in[0];
  // d_in[1] = batch_size (compile-time constant here)
  const float* dW   = (const float*)d_in[2];
  const float* drW0 = (const float*)d_in[3];
  const float* drb0 = (const float*)d_in[4];
  const float* drW1 = (const float*)d_in[5];
  const float* drb1 = (const float*)d_in[6];
  const float* drWo = (const float*)d_in[7];
  const float* drbo = (const float*)d_in[8];
  const float* dfW0 = (const float*)d_in[9];
  const float* dfb0 = (const float*)d_in[10];
  const float* dfW1 = (const float*)d_in[11];
  const float* dfb1 = (const float*)d_in[12];
  const float* dfWo = (const float*)d_in[13];
  const float* dfbo = (const float*)d_in[14];
  const float* roW  = (const float*)d_in[15];
  const float* rob  = (const float*)d_in[16];

  _Float16* wsh = (_Float16*)d_ws;
  float* out = (float*)d_out;

  sde_prep<<<64, 256, 0, stream>>>(drW0, drW1, drWo, dfW0, dfW1, dfWo, roW, wsh);
  sde_main<<<BATCH / 16, 128, 0, stream>>>(ts, dW,
                                           drW0, drb0, drb1, drbo,
                                           dfW0, dfb0, dfb1, dfbo,
                                           rob, wsh, out);
  (void)in_sizes; (void)n_in; (void)out_size; (void)ws_size;
}
